// TensorNetworkAttention_15848429322831
// MI455X (gfx1250) — matrix operand order repaired and checked by reading, not compiled, not run on hardware
//
#include <hip/hip_runtime.h>

#define B_DIM 4
#define SEQ   2048
#define H_DIM 4
#define DH    64
#define DM    256
#define ROWS  (B_DIM * SEQ)   // 8192

typedef __attribute__((ext_vector_type(16))) __bf16         bf16x16;
typedef __attribute__((ext_vector_type(8)))  float          f32x8;
typedef __attribute__((ext_vector_type(4)))  float          f32x4;
typedef __attribute__((ext_vector_type(8)))  unsigned short u16x8;

union BF16Vec {
    bf16x16        v;
    u16x8          h[2];
    unsigned short u[16];
};

__device__ __forceinline__ unsigned short f2bf(float f) {
    union { float f; unsigned int u; } c;
    c.f = f;
    unsigned int u = c.u;
    u += 0x7FFFu + ((u >> 16) & 1u);   // round-to-nearest-even
    return (unsigned short)(u >> 16);
}

__device__ __forceinline__ f32x8 wmma_bf16(bf16x16 a, bf16x16 b, f32x8 c) {
    // v_wmma_f32_16x16x32_bf16 : D = A(16x32) * B(32x16) + C(16x16 f32)
    return __builtin_amdgcn_wmma_f32_16x16x32_bf16(false, a, false, b,
                                                   (short)0, c, false, false);
}

// lane <-> lane^16 exchange (VALU v_permlanex16_b32 if available, else DS path)
__device__ __forceinline__ float xhalf(float x) {
#if __has_builtin(__builtin_amdgcn_permlanex16)
    int s = __builtin_bit_cast(int, x);
    int r = __builtin_amdgcn_permlanex16(s, s, 0x76543210, 0xfedcba98, false, false);
    return __builtin_bit_cast(float, r);
#else
    return __shfl_xor(x, 16, 32);
#endif
}

// ---------------------------------------------------------------------------
// Kernel 1: fold the tensor-interaction into Wq.
//   E_h = I + (alpha/64) * sum_r W2[h,:,:,r] @ W2[h,:,:,r]      (64x64)
//   Wq_eff[:, h*64+a] = 0.125 * sum_d Wq[:, h*64+d] * E_h[d,a]  (1/8 = softmax scale)
// ---------------------------------------------------------------------------
__global__ void prep_kernel(const float* __restrict__ Wq,
                            const float* __restrict__ bq,
                            const float* __restrict__ W2,
                            const float* __restrict__ alpha_p,
                            float* __restrict__ wq_eff,
                            float* __restrict__ bq_eff) {
    __shared__ float E[DH * DH];
    const int h   = blockIdx.x;
    const int tid = threadIdx.x;
    const float c = alpha_p[0] * (1.0f / (float)DH);

    for (int idx = tid; idx < DH * DH; idx += 256) {
        int d = idx >> 6, e = idx & 63;
        float s = 0.0f;
        for (int a = 0; a < DH; a++) {
            const float* p0 = W2 + (((size_t)(h * DH + d) * DH + a) * 4);
            const float* p1 = W2 + (((size_t)(h * DH + a) * DH + e) * 4);
            #pragma unroll
            for (int r = 0; r < 4; r++) s += p0[r] * p1[r];
        }
        E[idx] = ((d == e) ? 1.0f : 0.0f) + c * s;
    }
    __syncthreads();

    for (int idx = tid; idx < DM * DH; idx += 256) {
        int row = idx >> 6, a = idx & 63;
        float s = 0.0f;
        const float* wr = Wq + (size_t)row * DM + h * DH;
        for (int d = 0; d < DH; d++) s += wr[d] * E[d * DH + a];
        wq_eff[(size_t)row * DM + h * DH + a] = 0.125f * s;
    }
    if (tid < DH) {
        int a = tid;
        float s = 0.0f;
        for (int d = 0; d < DH; d++) s += bq[h * DH + d] * E[d * DH + a];
        bq_eff[h * DH + a] = 0.125f * s;
    }
}

// ---------------------------------------------------------------------------
// Kernel 2: one-shot weight convert+transpose: Wt[n][k] = bf16(W[k][n])
// grid: 256 blocks x 256 threads per matrix
// ---------------------------------------------------------------------------
__global__ void wconv_kernel(const float* __restrict__ W,
                             unsigned short* __restrict__ Wt) {
    const int n = blockIdx.x;
    const int k = threadIdx.x;
    Wt[(size_t)n * DM + k] = f2bf(W[(size_t)k * DM + n]);
}

// ---------------------------------------------------------------------------
// Kernel 3: WMMA GEMM projection   out = X(8192x256) @ W(256x256) + bias
//   W pre-converted bf16, transposed [n][k] -> B operand is 2 contiguous b128s.
//   AMODE: 0 = X f32, 1 = X bf16
//   OMODE: 0 = bf16 row-major; 1 = bf16 V-transposed [b][h][d][s]; 2 = f32
// grid: 512 blocks x 128 threads (wave = 16x64 output tile)
// ---------------------------------------------------------------------------
template <int AMODE, int OMODE>
__global__ void proj_kernel(const void* __restrict__ Xv,
                            const unsigned short* __restrict__ Wt,
                            const float* __restrict__ bias,
                            void* __restrict__ outv) {
    const int wid  = threadIdx.x >> 5;
    const int lane = threadIdx.x & 31;
    const int hi   = lane >> 4;
    const int ln   = lane & 15;
    const int w    = blockIdx.x * 4 + wid;
    const int row0 = (w >> 2) * 16;
    const int col0 = (w & 3) * 64;

    f32x8 acc[4];
    #pragma unroll
    for (int j = 0; j < 4; j++)
        #pragma unroll
        for (int v = 0; v < 8; v++) acc[j][v] = 0.0f;

    for (int ks = 0; ks < 8; ks++) {
        const int k0 = ks * 32;
        BF16Vec A;
        if (AMODE == 0) {
            const float* xr = (const float*)Xv + (size_t)(row0 + ln) * DM + k0;
            #pragma unroll
            for (int g = 0; g < 2; g++) {
                const float* p = xr + 16 * g + 8 * hi;
                f32x4 a0 = *(const f32x4*)p;
                f32x4 a1 = *(const f32x4*)(p + 4);
                #pragma unroll
                for (int i = 0; i < 4; i++) {
                    A.u[g * 8 + i]     = f2bf(a0[i]);
                    A.u[g * 8 + 4 + i] = f2bf(a1[i]);
                }
            }
        } else {
            const unsigned short* xr =
                (const unsigned short*)Xv + (size_t)(row0 + ln) * DM + k0;
            A.h[0] = *(const u16x8*)(xr + 8 * hi);
            A.h[1] = *(const u16x8*)(xr + 16 + 8 * hi);
        }
        #pragma unroll
        for (int j = 0; j < 4; j++) {
            const unsigned short* wp =
                Wt + (size_t)(col0 + 16 * j + ln) * DM + k0 + 8 * hi;
            BF16Vec Bv;
            Bv.h[0] = *(const u16x8*)(wp);
            Bv.h[1] = *(const u16x8*)(wp + 16);
            acc[j] = wmma_bf16(A.v, Bv.v, acc[j]);
        }
    }

    #pragma unroll
    for (int j = 0; j < 4; j++) {
        const int c     = col0 + 16 * j + ln;
        const float bvv = bias[c];
        #pragma unroll
        for (int v = 0; v < 8; v++) {
            const float o = acc[j][v] + bvv;
            const int row = row0 + v + 8 * hi;
            if (OMODE == 0) {
                ((unsigned short*)outv)[(size_t)row * DM + c] = f2bf(o);
            } else if (OMODE == 1) {
                const int hh = c >> 6, dd = c & 63;
                const int bb = row >> 11, s = row & (SEQ - 1);
                ((unsigned short*)outv)[((size_t)(bb * H_DIM + hh) * DH + dd) * SEQ + s] = f2bf(o);
            } else {
                ((float*)outv)[(size_t)row * DM + c] = o;
            }
        }
    }
}

// ---------------------------------------------------------------------------
// Kernel 4: flash attention, fully transposed formulation.
//   S^T = K * Q^T  -> each lane owns ONE query: softmax stats are per-lane
//   O^T = V^T * P^T -> P^T B-operand assembled in registers (8 permlanex16)
//   No LDS, no barriers, waves fully independent.
// grid: 512 blocks x 128 threads (wave = one 16-query tile of one (b,h))
// ---------------------------------------------------------------------------
__global__ void flash_kernel(const unsigned short* __restrict__ qbf,
                             const unsigned short* __restrict__ kbf,
                             const unsigned short* __restrict__ vt,
                             unsigned short* __restrict__ obf) {
    const int wid  = threadIdx.x >> 5;
    const int lane = threadIdx.x & 31;
    const int hi   = lane >> 4;
    const int ln   = lane & 15;
    const int w    = blockIdx.x * 4 + wid;  // 0..2047
    const int b    = w >> 9;
    const int h    = (w >> 7) & 3;
    const int q0   = (w & 127) * 16;

    const unsigned short* qrow  = qbf + (size_t)(b * SEQ + q0 + ln) * DM + h * DH;
    const unsigned short* kbase = kbf + (size_t)(b * SEQ) * DM + h * DH;
    const unsigned short* vbase = vt + (size_t)((b * H_DIM + h) * DH) * SEQ;

    // Q as B-operand (Q^T): B[k=d][n=query], two 16B chunks (d=8hi+0..7, 16+8hi+0..7) per lane per kstep
    BF16Vec Qb[2];
    #pragma unroll
    for (int ks = 0; ks < 2; ks++) {
        Qb[ks].h[0] = *(const u16x8*)(qrow + 32 * ks + 8 * hi);
        Qb[ks].h[1] = *(const u16x8*)(qrow + 32 * ks + 16 + 8 * hi);
    }

    f32x8 acc[4];
    #pragma unroll
    for (int j = 0; j < 4; j++)
        #pragma unroll
        for (int v = 0; v < 8; v++) acc[j][v] = 0.0f;
    float mrun = -1.0e30f, lrun = 0.0f;   // per-lane: stats for query q0+ln

    for (int kc = 0; kc < SEQ; kc += 32) {
        if (kc + 32 < SEQ) {  // uniform branch; prefetch next chunk
            __builtin_prefetch(kbase + (size_t)(kc + 32 + ln) * DM, 0, 1);
            __builtin_prefetch(vbase + (size_t)ln * SEQ + kc + 32, 0, 1);
        }
        // S^T tiles: St[t][v] = score(key kc+16t+v+8*hi, query q0+ln)
        f32x8 St[2];
        #pragma unroll
        for (int t = 0; t < 2; t++) {
            #pragma unroll
            for (int v = 0; v < 8; v++) St[t][v] = 0.0f;
            const unsigned short* krow = kbase + (size_t)(kc + t * 16 + ln) * DM;
            #pragma unroll
            for (int ks = 0; ks < 2; ks++) {
                BF16Vec Ka;
                Ka.h[0] = *(const u16x8*)(krow + 32 * ks + 8 * hi);
                Ka.h[1] = *(const u16x8*)(krow + 32 * ks + 16 + 8 * hi);
                St[t] = wmma_bf16(Ka.v, Qb[ks].v, St[t]);
            }
        }
        // per-lane online softmax over this lane's 16 keys + partner's 16
        float mloc = fmaxf(St[0][0], St[1][0]);
        #pragma unroll
        for (int v = 1; v < 8; v++)
            mloc = fmaxf(mloc, fmaxf(St[0][v], St[1][v]));
        mloc = fmaxf(mloc, xhalf(mloc));
        const float mnew = fmaxf(mrun, mloc);
        const float corr = __expf(mrun - mnew);
        mrun = mnew;
        float ssum = 0.0f;
        #pragma unroll
        for (int t = 0; t < 2; t++)
            #pragma unroll
            for (int v = 0; v < 8; v++) {
                const float p = __expf(St[t][v] - mnew);
                St[t][v] = p;
                ssum += p;
            }
        ssum += xhalf(ssum);
        lrun = lrun * corr + ssum;
        #pragma unroll
        for (int j = 0; j < 4; j++)
            #pragma unroll
            for (int v = 0; v < 8; v++) acc[j][v] *= corr;   // per-lane (per-query)

        // assemble P^T B-operand (32 keys x 16 queries) via lane^16 exchange
        float lowv[8], highv[8];
        #pragma unroll
        for (int v = 0; v < 8; v++) {
            const float send  = hi ? St[0][v] : St[1][v];   // what partner needs
            const float cross = xhalf(send);
            lowv[v]  = hi ? cross    : St[0][v];   // rows 0..7  of tile `hi`
            highv[v] = hi ? St[1][v] : cross;      // rows 8..15 of tile `hi`
        }
        BF16Vec Pb;
        #pragma unroll
        for (int v = 0; v < 8; v++) {
            Pb.u[v]     = f2bf(lowv[v]);
            Pb.u[8 + v] = f2bf(highv[v]);
        }
        // O^T += V^T * P^T   (V^T A-operands: contiguous from pre-transposed V)
        #pragma unroll
        for (int j = 0; j < 4; j++) {
            const unsigned short* vrow =
                vbase + (size_t)(16 * j + ln) * SEQ + kc;
            BF16Vec Va;
            Va.h[0] = *(const u16x8*)(vrow + 16 * hi);
            Va.h[1] = *(const u16x8*)(vrow + 16 * hi + 8);
            acc[j] = wmma_bf16(Va.v, Pb.v, acc[j]);
        }
    }

    // epilogue: acc[j] row v+8*hi = O^T[d = 16j+v+8hi][query = ln]
    const float inv = 1.0f / lrun;
    unsigned short* orow = obf + (size_t)(b * SEQ + q0 + ln) * DM + h * DH;
    #pragma unroll
    for (int j = 0; j < 4; j++) {
        u16x8 pack;
        #pragma unroll
        for (int v = 0; v < 8; v++) pack[v] = f2bf(acc[j][v] * inv);
        *(u16x8*)(orow + 16 * j + 8 * hi) = pack;
    }
}

// ---------------------------------------------------------------------------
// Workspace layout (bytes)
// ---------------------------------------------------------------------------
static constexpr size_t WQE_OFF = 0;                        // 256KB f32 Wq_eff
static constexpr size_t BQE_OFF = 262144;                   // 1KB   f32 bq_eff
static constexpr size_t WTQ_OFF = 263168;                   // 128KB bf16 Wq^T
static constexpr size_t WTK_OFF = WTQ_OFF + 131072;
static constexpr size_t WTV_OFF = WTK_OFF + 131072;
static constexpr size_t WTO_OFF = WTV_OFF + 131072;
static constexpr size_t QBF_OFF = WTO_OFF + 131072;
static constexpr size_t KBF_OFF = QBF_OFF + (size_t)ROWS * DM * 2;
static constexpr size_t VT_OFF  = KBF_OFF + (size_t)ROWS * DM * 2;
static constexpr size_t OBF_OFF = VT_OFF  + (size_t)ROWS * DM * 2;

extern "C" void kernel_launch(void* const* d_in, const int* in_sizes, int n_in,
                              void* d_out, int out_size, void* d_ws, size_t ws_size,
                              hipStream_t stream) {
    (void)in_sizes; (void)n_in; (void)out_size; (void)ws_size;
    const float* query = (const float*)d_in[0];
    const float* key_i = (const float*)d_in[1];
    const float* value = (const float*)d_in[2];
    const float* Wq    = (const float*)d_in[3];
    const float* bq    = (const float*)d_in[4];
    const float* Wk    = (const float*)d_in[5];
    const float* bk    = (const float*)d_in[6];
    const float* Wv    = (const float*)d_in[7];
    const float* bv    = (const float*)d_in[8];
    const float* W2    = (const float*)d_in[9];
    const float* alpha = (const float*)d_in[10];
    const float* Wo    = (const float*)d_in[11];
    const float* bo    = (const float*)d_in[12];

    char* ws = (char*)d_ws;
    float*          wq_eff = (float*)(ws + WQE_OFF);
    float*          bq_eff = (float*)(ws + BQE_OFF);
    unsigned short* wtq    = (unsigned short*)(ws + WTQ_OFF);
    unsigned short* wtk    = (unsigned short*)(ws + WTK_OFF);
    unsigned short* wtv    = (unsigned short*)(ws + WTV_OFF);
    unsigned short* wto    = (unsigned short*)(ws + WTO_OFF);
    unsigned short* qbf    = (unsigned short*)(ws + QBF_OFF);
    unsigned short* kbf    = (unsigned short*)(ws + KBF_OFF);
    unsigned short* vt     = (unsigned short*)(ws + VT_OFF);
    unsigned short* obf    = (unsigned short*)(ws + OBF_OFF);

    prep_kernel<<<H_DIM, 256, 0, stream>>>(Wq, bq, W2, alpha, wq_eff, bq_eff);

    wconv_kernel<<<DM, DM, 0, stream>>>(wq_eff, wtq);
    wconv_kernel<<<DM, DM, 0, stream>>>(Wk, wtk);
    wconv_kernel<<<DM, DM, 0, stream>>>(Wv, wtv);
    wconv_kernel<<<DM, DM, 0, stream>>>(Wo, wto);

    proj_kernel<0, 0><<<512, 128, 0, stream>>>(query, wtq, bq_eff, qbf);
    proj_kernel<0, 0><<<512, 128, 0, stream>>>(key_i, wtk, bk, kbf);
    proj_kernel<0, 1><<<512, 128, 0, stream>>>(value, wtv, bv, vt);

    flash_kernel<<<512, 128, 0, stream>>>(qbf, kbf, vt, obf);

    proj_kernel<1, 2><<<512, 128, 0, stream>>>(obf, wto, bo, d_out);
}
